// SelfAttention_3736621548226
// MI455X (gfx1250) — compile-verified, matrix operand order checked by reading, not run on hardware
//
#include <hip/hip_runtime.h>
#include <math.h>

typedef __attribute__((ext_vector_type(16))) __bf16 v16bf;
typedef __attribute__((ext_vector_type(8)))  __bf16 v8bf;
typedef __attribute__((ext_vector_type(8)))  float  v8f;

union V16 { v16bf v; v8bf h[2]; };

constexpr int kB  = 2;
constexpr int kN  = 2048;
constexpr int kD  = 1024;
constexpr int kH  = 16;
constexpr int kHS = 64;
constexpr int k3D = 3 * kD;
constexpr int kM  = kB * kN;          // 4096 rows of x

// ---- gfx1250 async global->LDS path (guarded; falls back to sync staging) ----
#if defined(__gfx1250__) && __has_builtin(__builtin_amdgcn_global_load_async_to_lds_b128)
#define ASYNC_LDS 1
typedef int v4i_vec __attribute__((vector_size(16)));
typedef __attribute__((address_space(1))) v4i_vec as1_v4i;
typedef __attribute__((address_space(3))) v4i_vec as3_v4i;
#else
#define ASYNC_LDS 0
#endif

__device__ inline void stage_copy16(const __bf16* src, __bf16* dst) {
#if ASYNC_LDS
  __builtin_amdgcn_global_load_async_to_lds_b128(
      (as1_v4i*)(uintptr_t)src, (as3_v4i*)(uintptr_t)dst, 0, 0);
#else
  *(v8bf*)dst = *(const v8bf*)src;
#endif
}

__device__ inline void stage_fence() {
#if ASYNC_LDS
#if __has_builtin(__builtin_amdgcn_s_wait_asynccnt)
  __builtin_amdgcn_s_wait_asynccnt(0);
#else
  asm volatile("s_wait_asynccnt 0x0" ::: "memory");
#endif
#endif
}

__device__ inline v16bf load_frag(const __bf16* p0, const __bf16* p1) {
  V16 u;
  u.h[0] = *(const v8bf*)p0;
  u.h[1] = *(const v8bf*)p1;
  return u.v;
}

__device__ inline v8f wmma_bf16(v16bf a, v16bf b, v8f c) {
  return __builtin_amdgcn_wmma_f32_16x16x32_bf16(false, a, false, b, (short)0, c,
                                                 false, false);
}

// ---------------- conversion kernels ----------------
__global__ void cvt_f32_bf16(const float* __restrict__ in, __bf16* __restrict__ out, int n) {
  int i = blockIdx.x * blockDim.x + threadIdx.x;
  if (i < n) out[i] = (__bf16)in[i];
}

// in: [rows][cols] f32 row-major  ->  out: [cols][rows] bf16 (i.e. transposed)
__global__ void transpose_f32_bf16(const float* __restrict__ in, __bf16* __restrict__ out,
                                   int rows, int cols) {
  int i = blockIdx.x * blockDim.x + threadIdx.x;
  if (i < rows * cols) {
    int r = i / cols, c = i % cols;
    out[c * rows + r] = (__bf16)in[i];
  }
}

// ---------------- QKV projection GEMM ----------------
// Wave tile 32x64, double-buffered fragments + L2 prefetch two k-steps ahead.
__global__ void __launch_bounds__(256) gemm_qkv(const __bf16* __restrict__ A,
                                                const __bf16* __restrict__ Bt,
                                                __bf16* __restrict__ Q,
                                                __bf16* __restrict__ Kb,
                                                __bf16* __restrict__ Vt) {
  const int lane = threadIdx.x & 31;
  const int wave = threadIdx.x >> 5;
  const int t    = blockIdx.x * 8 + wave;   // 128 row tiles * 48 col tiles
  const int rt   = t / 48, ct = t % 48;
  const int m0   = rt * 32, c0 = ct * 64;
  const int mrow = lane & 15;
  const int half = lane >> 4;
  const int h8   = half * 8;

  const v8f zero = {0.f, 0.f, 0.f, 0.f, 0.f, 0.f, 0.f, 0.f};
  v8f acc[2][4] = {{zero, zero, zero, zero}, {zero, zero, zero, zero}};
  v16bf aF[2][2], bF[2][4];

  auto loadA = [&](int k0, v16bf a[2]) {
#pragma unroll
    for (int ra = 0; ra < 2; ++ra) {
      const __bf16* ar = A + (size_t)(m0 + ra * 16 + mrow) * kD + k0;
      a[ra] = load_frag(ar + h8, ar + 16 + h8);
    }
  };
  auto loadB = [&](int k0, v16bf bb[4]) {
#pragma unroll
    for (int j = 0; j < 4; ++j) {
      const __bf16* bp = Bt + (size_t)(c0 + j * 16 + mrow) * kD + k0;
      bb[j] = load_frag(bp + h8, bp + 16 + h8);
    }
  };

  loadA(0, aF[0]);
  loadB(0, bF[0]);
  int cur = 0;
  for (int k0 = 0; k0 < kD; k0 += 32) {
    int nxt = cur ^ 1;
    if (k0 + 32 < kD) { loadA(k0 + 32, aF[nxt]); loadB(k0 + 32, bF[nxt]); }
    if (k0 + 64 < kD) {
      __builtin_prefetch(A + (size_t)(m0 + mrow) * kD + k0 + 64, 0, 1);
      __builtin_prefetch(Bt + (size_t)(c0 + mrow) * kD + k0 + 64, 0, 1);
    }
#pragma unroll
    for (int ra = 0; ra < 2; ++ra)
#pragma unroll
      for (int j = 0; j < 4; ++j)
        acc[ra][j] = wmma_bf16(aF[cur][ra], bF[cur][j], acc[ra][j]);
    cur = nxt;
  }

  const int sec = c0 / kD;            // 0=q 1=k 2=v
  const int hh  = (c0 % kD) / kHS;    // head (tile never crosses a head)
#pragma unroll
  for (int ra = 0; ra < 2; ++ra) {
#pragma unroll
    for (int j = 0; j < 4; ++j) {
#pragma unroll
      for (int v = 0; v < 8; ++v) {
        int m = m0 + ra * 16 + v + half * 8;
        int b = m >> 11, n = m & (kN - 1);
        int e = j * 16 + mrow;
        __bf16 bv = (__bf16)acc[ra][j][v];
        size_t qk_idx = (((size_t)(b * kH + hh) * kN + n) * kHS) + e;
        if (sec == 0)      Q[qk_idx]  = bv;
        else if (sec == 1) Kb[qk_idx] = bv;
        else               Vt[(((size_t)(b * kH + hh) * kHS + e) * kN) + n] = bv;
      }
    }
  }
}

// ---------------- flash attention (workgroup-cooperative) ----------------
// One workgroup = one (b,h) and 8 query tiles (one per wave).  The 32-key
// K/V block is staged into LDS once per workgroup, double-buffered, using
// the gfx1250 async global->LDS engine when available.
__global__ void __launch_bounds__(256) attn(const __bf16* __restrict__ Q,
                                            const __bf16* __restrict__ Kb,
                                            const __bf16* __restrict__ Vt,
                                            const float* __restrict__ mask,
                                            __bf16* __restrict__ ctx) {
  __shared__ __attribute__((aligned(16))) __bf16 ldsK[2][32][64];  // [key][dim]
  __shared__ __attribute__((aligned(16))) __bf16 ldsV[2][64][32];  // [dim][key]
  __shared__ __attribute__((aligned(16))) __bf16 ldsP[8][16][32];

  const int tid  = threadIdx.x;
  const int lane = tid & 31;
  const int wave = tid >> 5;
  const int bh   = blockIdx.x / 16;           // 32 (b,h) pairs
  const int qg   = blockIdx.x % 16;           // query-tile group
  const int b    = bh / kH;
  const int h    = bh % kH;
  const int n0   = (qg * 8 + wave) * 16;
  const int mrow = lane & 15;
  const int half = lane >> 4;
  const int h8   = half * 8;
  const float scale = 0.03125f;               // full d^-0.5, faithful to module

  const __bf16* Qbase = Q  + (size_t)bh * kN * kHS;
  const __bf16* Kbase = Kb + (size_t)bh * kN * kHS;
  const __bf16* Vbase = Vt + (size_t)bh * kHS * kN;

  const __bf16* qp = Qbase + (size_t)(n0 + mrow) * kHS;
  v16bf aQ0 = load_frag(qp + h8,      qp + 16 + h8);   // head-dims 0..31
  v16bf aQ1 = load_frag(qp + 32 + h8, qp + 48 + h8);   // head-dims 32..63

  float mq[8];
#pragma unroll
  for (int v = 0; v < 8; ++v) mq[v] = mask[b * kN + n0 + v + half * 8];

  const v8f zero = {0.f, 0.f, 0.f, 0.f, 0.f, 0.f, 0.f, 0.f};
  v8f cacc[4] = {zero, zero, zero, zero};
  float rmax[8], rsum[8];
#pragma unroll
  for (int v = 0; v < 8; ++v) { rmax[v] = -__builtin_inff(); rsum[v] = 0.f; }

  // cooperative staging: each thread moves one 16B chunk of K and one of V
  const int kKey   = tid >> 3;         // 0..31
  const int kChunk = (tid & 7) * 8;    // 0..56
  const int vRow   = tid >> 2;         // 0..63
  const int vChunk = (tid & 3) * 8;    // 0..24
  auto stage = [&](int bufi, int kb) {
    stage_copy16(Kbase + (size_t)(kb + kKey) * kHS + kChunk,
                 &ldsK[bufi][kKey][kChunk]);
    stage_copy16(Vbase + (size_t)vRow * kN + kb + vChunk,
                 &ldsV[bufi][vRow][vChunk]);
  };

  stage(0, 0);
  stage_fence();
  __syncthreads();
  int buf = 0;

  for (int kb = 0; kb < kN; kb += 32) {
    if (kb + 32 < kN) stage(buf ^ 1, kb + 32);   // overlap with compute

    float mk0 = mask[b * kN + kb + mrow];
    float mk1 = mask[b * kN + kb + 16 + mrow];

    v16bf bK00 = load_frag(&ldsK[buf][mrow][h8],           &ldsK[buf][mrow][16 + h8]);
    v16bf bK01 = load_frag(&ldsK[buf][mrow][32 + h8],      &ldsK[buf][mrow][48 + h8]);
    v16bf bK10 = load_frag(&ldsK[buf][16 + mrow][h8],      &ldsK[buf][16 + mrow][16 + h8]);
    v16bf bK11 = load_frag(&ldsK[buf][16 + mrow][32 + h8], &ldsK[buf][16 + mrow][48 + h8]);
    v8f s0 = zero, s1 = zero;
    s0 = wmma_bf16(aQ0, bK00, s0);
    s0 = wmma_bf16(aQ1, bK01, s0);
    s1 = wmma_bf16(aQ0, bK10, s1);
    s1 = wmma_bf16(aQ1, bK11, s1);

    // issue V-fragment loads early; softmax VALU work below hides them
    v16bf bV[4];
#pragma unroll
    for (int j = 0; j < 4; ++j)
      bV[j] = load_frag(&ldsV[buf][j * 16 + mrow][h8],
                        &ldsV[buf][j * 16 + mrow][16 + h8]);

    float p0[8], p1[8];
#pragma unroll
    for (int v = 0; v < 8; ++v) {
      float sv0 = s0[v] * scale * mq[v] * mk0;
      float sv1 = s1[v] * scale * mq[v] * mk1;
      if (sv0 == 0.0f) sv0 = -__builtin_inff();  // masked -> -inf (ref semantics)
      if (sv1 == 0.0f) sv1 = -__builtin_inff();
      float tm = fmaxf(sv0, sv1);                // row lives in one 16-lane half
      tm = fmaxf(tm, __shfl_xor(tm, 1, 32));
      tm = fmaxf(tm, __shfl_xor(tm, 2, 32));
      tm = fmaxf(tm, __shfl_xor(tm, 4, 32));
      tm = fmaxf(tm, __shfl_xor(tm, 8, 32));
      float mnew = fmaxf(rmax[v], tm);
      float fac  = (rmax[v] == mnew) ? 1.0f : __expf(rmax[v] - mnew);
      float e0 = __expf(sv0 - mnew);
      float e1 = __expf(sv1 - mnew);
      float ts = e0 + e1;
      ts += __shfl_xor(ts, 1, 32);
      ts += __shfl_xor(ts, 2, 32);
      ts += __shfl_xor(ts, 4, 32);
      ts += __shfl_xor(ts, 8, 32);
      rsum[v] = rsum[v] * fac + ts;
      rmax[v] = mnew;
      p0[v] = e0; p1[v] = e1;
#pragma unroll
      for (int j = 0; j < 4; ++j) cacc[j][v] *= fac;
    }

    // redistribute P from C/D layout to A-fragment layout via LDS (per-wave)
#pragma unroll
    for (int v = 0; v < 8; ++v) {
      int r = v + half * 8;
      ldsP[wave][r][mrow]      = (__bf16)p0[v];
      ldsP[wave][r][16 + mrow] = (__bf16)p1[v];
    }
    asm volatile("s_wait_dscnt 0" ::: "memory");
    v16bf aP = load_frag(&ldsP[wave][mrow][h8], &ldsP[wave][mrow][16 + h8]);

#pragma unroll
    for (int j = 0; j < 4; ++j) cacc[j] = wmma_bf16(aP, bV[j], cacc[j]);

    stage_fence();     // this thread's async copies into buf^1 have landed
    __syncthreads();   // publish to all waves; buf free for next overwrite
    buf ^= 1;
  }

#pragma unroll
  for (int v = 0; v < 8; ++v) {
    float inv = 1.0f / rsum[v];
    int n = n0 + v + half * 8;
#pragma unroll
    for (int j = 0; j < 4; ++j) {
      ctx[((size_t)b * kN + n) * kD + h * kHS + j * 16 + mrow] =
          (__bf16)(cacc[j][v] * inv);
    }
  }
}

// ---------------- output projection GEMM (+bias, f32 out) ----------------
__global__ void __launch_bounds__(256) gemm_out(const __bf16* __restrict__ A,
                                                const __bf16* __restrict__ Bt,
                                                const float* __restrict__ bias,
                                                float* __restrict__ out) {
  const int lane = threadIdx.x & 31;
  const int wave = threadIdx.x >> 5;
  const int t    = blockIdx.x * 8 + wave;   // 128 row tiles * 16 col tiles
  const int rt   = t / 16, ct = t % 16;
  const int m0   = rt * 32, c0 = ct * 64;
  const int mrow = lane & 15;
  const int half = lane >> 4;
  const int h8   = half * 8;

  const v8f zero = {0.f, 0.f, 0.f, 0.f, 0.f, 0.f, 0.f, 0.f};
  v8f acc[2][4] = {{zero, zero, zero, zero}, {zero, zero, zero, zero}};
  v16bf aF[2][2], bF[2][4];

  auto loadA = [&](int k0, v16bf a[2]) {
#pragma unroll
    for (int ra = 0; ra < 2; ++ra) {
      const __bf16* ar = A + (size_t)(m0 + ra * 16 + mrow) * kD + k0;
      a[ra] = load_frag(ar + h8, ar + 16 + h8);
    }
  };
  auto loadB = [&](int k0, v16bf bb[4]) {
#pragma unroll
    for (int j = 0; j < 4; ++j) {
      const __bf16* bp = Bt + (size_t)(c0 + j * 16 + mrow) * kD + k0;
      bb[j] = load_frag(bp + h8, bp + 16 + h8);
    }
  };

  loadA(0, aF[0]);
  loadB(0, bF[0]);
  int cur = 0;
  for (int k0 = 0; k0 < kD; k0 += 32) {
    int nxt = cur ^ 1;
    if (k0 + 32 < kD) { loadA(k0 + 32, aF[nxt]); loadB(k0 + 32, bF[nxt]); }
    if (k0 + 64 < kD) {
      __builtin_prefetch(A + (size_t)(m0 + mrow) * kD + k0 + 64, 0, 1);
      __builtin_prefetch(Bt + (size_t)(c0 + mrow) * kD + k0 + 64, 0, 1);
    }
#pragma unroll
    for (int ra = 0; ra < 2; ++ra)
#pragma unroll
      for (int j = 0; j < 4; ++j)
        acc[ra][j] = wmma_bf16(aF[cur][ra], bF[cur][j], acc[ra][j]);
    cur = nxt;
  }

#pragma unroll
  for (int ra = 0; ra < 2; ++ra) {
#pragma unroll
    for (int j = 0; j < 4; ++j) {
#pragma unroll
      for (int v = 0; v < 8; ++v) {
        int m = m0 + ra * 16 + v + half * 8;
        int c = c0 + j * 16 + mrow;
        out[(size_t)m * kD + c] = acc[ra][j][v] + bias[c];
      }
    }
  }
}

// ---------------- launcher ----------------
extern "C" void kernel_launch(void* const* d_in, const int* in_sizes, int n_in,
                              void* d_out, int out_size, void* d_ws, size_t ws_size,
                              hipStream_t stream) {
  const float* x    = (const float*)d_in[0];
  const float* mask = (const float*)d_in[1];
  const float* Wqkv = (const float*)d_in[2];
  const float* Wout = (const float*)d_in[3];
  const float* bout = (const float*)d_in[4];
  float* out = (float*)d_out;

  char* ws = (char*)d_ws;
  __bf16* xb    = (__bf16*)(ws);                 //  8 MB  x in bf16
  __bf16* WqkvT = (__bf16*)(ws + 8388608);       //  6 MB  Wqkv^T bf16
  __bf16* WoutT = (__bf16*)(ws + 14680064);      //  2 MB  Wout^T bf16
  __bf16* Qb    = (__bf16*)(ws + 16777216);      //  8 MB  [b,h,n,e]
  __bf16* Kbuf  = (__bf16*)(ws + 25165824);      //  8 MB  [b,h,n,e]
  __bf16* Vt    = (__bf16*)(ws + 33554432);      //  8 MB  [b,h,e,n]
  __bf16* ctx   = (__bf16*)(ws + 41943040);      //  8 MB  [b,n,d]

  cvt_f32_bf16<<<(kM * kD + 255) / 256, 256, 0, stream>>>(x, xb, kM * kD);
  transpose_f32_bf16<<<(kD * k3D + 255) / 256, 256, 0, stream>>>(Wqkv, WqkvT, kD, k3D);
  transpose_f32_bf16<<<(kD * kD + 255) / 256, 256, 0, stream>>>(Wout, WoutT, kD, kD);

  gemm_qkv<<<768, 256, 0, stream>>>(xb, WqkvT, Qb, Kbuf, Vt);    // 6144 waves
  attn<<<512, 256, 0, stream>>>(Qb, Kbuf, Vt, mask, ctx);        // 4096 waves
  gemm_out<<<256, 256, 0, stream>>>(ctx, WoutT, bout, out);      // 2048 waves
}
